// PointerNet_57011395887634
// MI455X (gfx1250) — matrix operand order repaired and checked by reading, not compiled, not run on hardware
//
#include <hip/hip_runtime.h>
#include <hip/hip_bf16.h>

#define B_ 4
#define T_ 256
#define I_ 512
#define H_ 8
#define V_ 32000
#define D_ 512

typedef _Float16 v16h __attribute__((ext_vector_type(16)));
typedef float    v8f  __attribute__((ext_vector_type(8)));
typedef float    f4   __attribute__((ext_vector_type(4)));

// ---------------------------------------------------------------------------
// Kernel 1: attn = mean over heads.  (B,H,T,I) -> (B,T,I), B128 vectorized.
// ---------------------------------------------------------------------------
__global__ __launch_bounds__(256)
void mean_heads_kernel(const float* __restrict__ ah, float* __restrict__ attn) {
    const int N4 = (B_ * T_ * I_) / 4;                  // 131072 float4's
    int idx = blockIdx.x * 256 + threadIdx.x;
    if (idx >= N4) return;
    const int TI4 = (T_ * I_) / 4;
    int b  = idx / TI4;
    int ti = idx - b * TI4;
    const f4* p = (const f4*)ah + (size_t)b * H_ * TI4 + ti;
    f4 s = {0.f, 0.f, 0.f, 0.f};
#pragma unroll
    for (int h = 0; h < H_; ++h) s += p[(size_t)h * TI4];
    ((f4*)attn)[idx] = s * (1.0f / H_);
}

// ---------------------------------------------------------------------------
// Kernel 2: context[b] (T x D) = attn[b] (T x I) @ enc[b] (I x D)
// One wave32 per 16x16 output tile, v_wmma_f32_16x16x32_f16, K-loop of 16.
// ---------------------------------------------------------------------------
__global__ __launch_bounds__(128)
void context_gemm_wmma(const float* __restrict__ attn,   // (B,T,I)
                       const float* __restrict__ enc,    // (B,I,D)
                       float* __restrict__ ctx)          // (B,T,D)
{
    const int lane = threadIdx.x & 31;
    const int wave = threadIdx.x >> 5;
    const int w    = blockIdx.x * 4 + wave;   // 0..2047 global wave id
    const int b    = w >> 9;                  // 512 tiles per batch
    const int tile = w & 511;
    const int t0   = (tile >> 5) << 4;        // tile row * 16   (T/16 = 16 rows)
    const int d0   = (tile & 31) << 4;        // tile col * 16   (D/16 = 32 cols)
    const int m    = lane & 15;               // also the N index for B/C frags
    const int half = lane >> 4;

    const float* Ab = attn + (size_t)b * T_ * I_;
    const float* Bb = enc  + (size_t)b * I_ * D_;

    v8f acc = {};
    for (int kb = 0; kb < I_; kb += 32) {
        v16h afrag, bfrag;
#pragma unroll
        for (int v = 0; v < 8; ++v) {
            // A-matrix 16x32 f16 layout: VGPR v, lane half ->
            //   k = (v/4)*16 + half*8 + (v%4)*2   (pair k, k+1)
            int ka = ((v >> 2) << 4) + (half << 3) + ((v & 3) << 1);
            const float* ap = Ab + (size_t)(t0 + m) * I_ + kb + ka;
            afrag[2 * v]     = (_Float16)ap[0];
            afrag[2 * v + 1] = (_Float16)ap[1];
            // B-matrix 32x16 f16 layout: lane n = lane%16, k = (v/4)*16 + half*8 + (v%4)*2
            int kB = ((v >> 2) << 4) + (half << 3) + ((v & 3) << 1);
            const float* bp = Bb + (size_t)(kb + kB) * D_ + d0 + m;
            bfrag[2 * v]     = (_Float16)bp[0];
            bfrag[2 * v + 1] = (_Float16)bp[D_];
        }
        acc = __builtin_amdgcn_wmma_f32_16x16x32_f16(
            /*neg_a=*/false, afrag, /*neg_b=*/false, bfrag,
            /*c_mod=*/(short)0, acc, /*reuse_a=*/false, /*reuse_b=*/false);
    }

    float* Cp = ctx + (size_t)b * T_ * D_;
#pragma unroll
    for (int r = 0; r < 8; ++r) {
        // C/D layout: VGPR r holds M = r + 8*half, N = lane%16
        Cp[(size_t)(t0 + r + 8 * half) * D_ + d0 + m] = acc[r];
    }
}

// ---------------------------------------------------------------------------
// Kernel 3: p_gen[b,t] = sigmoid([ctx|dec|tar] . w + bias).  One block per row.
// ---------------------------------------------------------------------------
__global__ __launch_bounds__(256)
void pgen_kernel(const float* __restrict__ ctx, const float* __restrict__ dec,
                 const float* __restrict__ tar, const float* __restrict__ w,
                 const float* __restrict__ bias,
                 float* __restrict__ pgen_ws, float* __restrict__ pgen_out) {
    const int row = blockIdx.x;                       // b*T + t
    const float* c  = ctx + (size_t)row * D_;
    const float* de = dec + (size_t)row * D_;
    const float* ta = tar + (size_t)row * D_;
    float s = 0.f;
    for (int d = threadIdx.x; d < D_; d += 256)
        s += c[d] * w[d] + de[d] * w[D_ + d] + ta[d] * w[2 * D_ + d];
    __shared__ float red[256];
    red[threadIdx.x] = s;
    __syncthreads();
    for (int off = 128; off > 0; off >>= 1) {
        if (threadIdx.x < off) red[threadIdx.x] += red[threadIdx.x + off];
        __syncthreads();
    }
    if (threadIdx.x == 0) {
        float z  = red[0] + bias[0];
        float pg = 1.f / (1.f + __expf(-z));
        pgen_ws[row]  = pg;
        pgen_out[row] = pg;
    }
}

// ---------------------------------------------------------------------------
// Kernel 4: fused scatter -> log_softmax -> scale -> blend.
// One block per (b,t) row; dense vocab row in LDS (125 KB of the WGP's 320KB).
// ds_add_f32 atomics dedupe repeated tokens.  The streaming pass over V uses
// B128 vector ops with non-temporal (TH_NT) cache hints: gen is read once and
// the 262 MB of output exceeds the 192 MB L2, so don't let it pollute caches.
// ---------------------------------------------------------------------------
__global__ __launch_bounds__(256)
void pointer_blend_kernel(const float* __restrict__ attn,   // (B,T,I)
                          const int*   __restrict__ tokens, // (B,I)
                          const float* __restrict__ gen,    // (B,T,V)
                          const float* __restrict__ pgen,   // (B*T)
                          const float* __restrict__ sw_p,
                          const float* __restrict__ sb_p,
                          float* __restrict__ out_final,
                          float* __restrict__ out_ptr) {
    __shared__ float row[V_];       // 128000 bytes
    __shared__ float red[256];
    const int rid = blockIdx.x;     // b*T + t
    const int b   = rid >> 8;       // rid / T_
    const float* a  = attn + (size_t)rid * I_;
    const int* tok  = tokens + (size_t)b * I_;
    const int V4 = V_ / 4;          // 8000

    f4* row4 = (f4*)row;
    const f4 zero4 = {0.f, 0.f, 0.f, 0.f};
    for (int v = threadIdx.x; v < V4; v += 256) row4[v] = zero4;   // ds_store_b128
    __syncthreads();

    for (int i = threadIdx.x; i < I_; i += 256)
        atomicAdd(&row[tok[i]], a[i]);          // ds_add_f32
    __syncthreads();

    // sum of exp over the dense row (values are probabilities <= 1; no max
    // subtraction needed for stability)
    float s = 0.f;
    for (int v = threadIdx.x; v < V4; v += 256) {
        f4 r = row4[v];                         // ds_load_b128
        s += __expf(r.x) + __expf(r.y) + __expf(r.z) + __expf(r.w);
    }
    red[threadIdx.x] = s;
    __syncthreads();
    for (int off = 128; off > 0; off >>= 1) {
        if (threadIdx.x < off) red[threadIdx.x] += red[threadIdx.x + off];
        __syncthreads();
    }
    const float logZ = __logf(red[0]);
    const float pg   = pgen[rid];
    const float sw   = sw_p[0];
    const float sb   = sb_p[0];

    const f4* g4  = (const f4*)(gen       + (size_t)rid * V_);
    f4*       of4 = (f4*)      (out_final + (size_t)rid * V_);
    f4*       op4 = (f4*)      (out_ptr   + (size_t)rid * V_);
    for (int v = threadIdx.x; v < V4; v += 256) {
        f4 r = row4[v];
        f4 g = __builtin_nontemporal_load(&g4[v]);      // global_load_b128 TH_NT
        f4 p = sw * (r - logZ) + sb;
        f4 f = pg * g + (1.f - pg) * p;
        __builtin_nontemporal_store(p, &op4[v]);        // global_store_b128 TH_NT
        __builtin_nontemporal_store(f, &of4[v]);
    }
}

// ---------------------------------------------------------------------------
// Host-side launcher
// Input order: inp_tokens, tar_embedded, generator_output, enc_output,
//              dec_state, attn_heads, p_gen_w, p_gen_b, pointer_scale_w,
//              pointer_scale_b
// Output: final (B,T,V) | pointer (B,T,V) | p_gen (B,T)   (float32)
// ---------------------------------------------------------------------------
extern "C" void kernel_launch(void* const* d_in, const int* in_sizes, int n_in,
                              void* d_out, int out_size, void* d_ws, size_t ws_size,
                              hipStream_t stream) {
    const int*   tokens = (const int*)  d_in[0];
    const float* tar    = (const float*)d_in[1];
    const float* gen    = (const float*)d_in[2];
    const float* enc    = (const float*)d_in[3];
    const float* dec    = (const float*)d_in[4];
    const float* ah     = (const float*)d_in[5];
    const float* w      = (const float*)d_in[6];
    const float* bias   = (const float*)d_in[7];
    const float* sw     = (const float*)d_in[8];
    const float* sb     = (const float*)d_in[9];

    float* ws_attn = (float*)d_ws;                    // B*T*I
    float* ws_ctx  = ws_attn + (size_t)B_ * T_ * I_;  // B*T*D
    float* ws_pgen = ws_ctx  + (size_t)B_ * T_ * D_;  // B*T

    float* out_final = (float*)d_out;
    float* out_ptr   = out_final + (size_t)B_ * T_ * V_;
    float* out_pg    = out_ptr   + (size_t)B_ * T_ * V_;

    mean_heads_kernel<<<(B_ * T_ * I_ / 4 + 255) / 256, 256, 0, stream>>>(ah, ws_attn);

    // 2048 tiles / 4 waves per block = 512 blocks of 128 threads
    context_gemm_wmma<<<512, 128, 0, stream>>>(ws_attn, enc, ws_ctx);

    pgen_kernel<<<B_ * T_, 256, 0, stream>>>(ws_ctx, dec, tar, w, bias,
                                             ws_pgen, out_pg);

    pointer_blend_kernel<<<B_ * T_, 256, 0, stream>>>(ws_attn, tokens, gen,
                                                      ws_pgen, sw, sb,
                                                      out_final, out_ptr);
}
